// TemporalEarthquakeGNN_18949395710306
// MI455X (gfx1250) — matrix operand order repaired and checked by reading, not compiled, not run on hardware
//
#include <hip/hip_runtime.h>
#include <hip/hip_bf16.h>
#include <math.h>

typedef __attribute__((ext_vector_type(16))) __bf16 v16bf;
typedef __attribute__((ext_vector_type(8)))  float  v8f;

#define N_NODES 100000
#define N_EDGES 3200000
#define HID     64
#define SEQ     10
#define NTILES  (N_NODES / 16)   // 6250 exact

static __device__ __forceinline__ v8f wmma_bf16(v16bf a, v16bf b, v8f c) {
  // D = A(16x32 bf16) * B(32x16 bf16) + C(16x16 f32)
  return __builtin_amdgcn_wmma_f32_16x16x32_bf16(
      /*neg_a=*/false, a, /*neg_b=*/false, b,
      /*c_mod=*/(short)0, c, /*reuse_a=*/false, /*reuse_b=*/false);
}
static __device__ __forceinline__ float sigf(float x) {
  return 1.0f / (1.0f + __expf(-x));
}

// ---------------------------------------------------------------- degree
__global__ __launch_bounds__(256) void deg_kernel(const int* __restrict__ dst,
                                                  float* __restrict__ deg) {
  int e = blockIdx.x * 256 + threadIdx.x;
  if (e < N_EDGES) atomicAdd(&deg[dst[e]], 1.0f);
}

__global__ __launch_bounds__(256) void dinv_kernel(const float* __restrict__ deg,
                                                   float* __restrict__ dinv) {
  int n = blockIdx.x * 256 + threadIdx.x;
  if (n < N_NODES) dinv[n] = rsqrtf(deg[n] + 1.0f);
}

// -------------------------------------------------- weight fragment prep
// B-fragment k order used here (contiguous, same as A): lane -> column N (lane%16), element i ->
// K = i + 16*(lane>=16) within a K=32 half. Fragment = 32 lanes x 16 bf16.
__global__ __launch_bounds__(256) void prep_kernel(
    const float* __restrict__ Wih, const float* __restrict__ Whh,
    const float* __restrict__ convW,
    const float* __restrict__ bih, const float* __restrict__ bhh,
    __bf16* __restrict__ wihF, __bf16* __restrict__ whhF,
    __bf16* __restrict__ convF, float* __restrict__ bsum) {
  int t = blockIdx.x * 256 + threadIdx.x;
  if (t < 1024) {                       // Wih^T fragments: 16 tiles x 2 khalf x 32 lanes
    int job = t, lane = job & 31, kh = (job >> 5) & 1, tile = job >> 6;
    int n = tile * 16 + (lane & 15);
    int kb = kh * 32 + ((lane >= 16) ? 16 : 0);
    for (int i = 0; i < 16; ++i)
      wihF[job * 16 + i] = (__bf16)Wih[n * HID + kb + i];   // B[k,n] = Wih[n,k]
  } else if (t < 2048) {                // Whh^T fragments
    int job = t - 1024, lane = job & 31, kh = (job >> 5) & 1, tile = job >> 6;
    int n = tile * 16 + (lane & 15);
    int kb = kh * 32 + ((lane >= 16) ? 16 : 0);
    for (int i = 0; i < 16; ++i)
      whhF[job * 16 + i] = (__bf16)Whh[n * HID + kb + i];
  } else if (t < 2560) {                // conv_W fragments: 2 layers x 4 tiles x 2 khalf
    int job = t - 2048, lane = job & 31, kh = (job >> 5) & 1;
    int tile = (job >> 6) & 3, layer = job >> 8;
    int n = tile * 16 + (lane & 15);
    int kb = kh * 32 + ((lane >= 16) ? 16 : 0);
    for (int i = 0; i < 16; ++i)        // B[k,n] = conv_W[layer][k][n]
      convF[job * 16 + i] = (__bf16)convW[layer * HID * HID + (kb + i) * HID + n];
  } else if (t < 2816) {
    int i = t - 2560;
    bsum[i] = bih[i] + bhh[i];
  }
}

// ------------------------------------------------------- input projection
__global__ __launch_bounds__(256) void inproj_kernel(
    const float* __restrict__ x, const float* __restrict__ Win,
    const float* __restrict__ bin, float* __restrict__ h) {
  int tid = blockIdx.x * 256 + threadIdx.x;      // N*64 threads
  int n = tid >> 6, j = tid & 63;
  float s = bin[j];
  const float* xr = x + n * 16;
#pragma unroll
  for (int k = 0; k < 16; ++k) s += xr[k] * Win[k * HID + j];
  h[tid] = fmaxf(s, 0.0f);
}

// ------------------------------------------------ GCN xt = h @ conv_W  (WMMA)
__global__ __launch_bounds__(256) void gcn_xt_kernel(
    const float* __restrict__ h, const __bf16* __restrict__ convFb,
    float* __restrict__ xt) {
  __shared__ __align__(16) __bf16 wl[4 * 2 * 32 * 16];   // 8KB layer fragments
  {
    const uint4* g = (const uint4*)convFb;
    uint4* l = (uint4*)wl;
    for (int i = threadIdx.x; i < 512; i += 256) l[i] = g[i];
  }
  __syncthreads();

  int w = threadIdx.x >> 5, lane = threadIdx.x & 31;
  int tile = blockIdx.x * 8 + w;
  bool valid = tile < NTILES;
  if (!valid) tile = NTILES - 1;
  int base = tile * 16, rlo = lane & 15, hi8 = (lane >= 16) ? 8 : 0;
  const v16bf* cf = (const v16bf*)wl;

  // A fragments (16x64 bf16 as two K=32 halves). Lane = row, elem i -> k.
  const float* rp = h + (size_t)(base + rlo) * HID;
  v16bf a0, a1;
#pragma unroll
  for (int i = 0; i < 8; ++i) {
    a0[i]     = (__bf16)rp[2 * hi8 + i];
    a0[i + 8] = (__bf16)rp[2 * hi8 + 8 + i];
    a1[i]     = (__bf16)rp[32 + 2 * hi8 + i];
    a1[i + 8] = (__bf16)rp[32 + 2 * hi8 + 8 + i];
  }
#pragma unroll
  for (int j = 0; j < 4; ++j) {
    v8f c = {};
    c = wmma_bf16(a0, cf[j * 64 + lane], c);
    c = wmma_bf16(a1, cf[j * 64 + 32 + lane], c);
    if (valid) {
#pragma unroll
      for (int r = 0; r < 8; ++r)
        xt[(size_t)(base + r + hi8) * HID + j * 16 + rlo] = c[r];
    }
  }
}

// -------------------------------------------- agg init: self-loop + bias
__global__ __launch_bounds__(256) void agg_init_kernel(
    const float* __restrict__ xt, const float* __restrict__ dinv,
    const float* __restrict__ cb, float* __restrict__ agg) {
  int tid = blockIdx.x * 256 + threadIdx.x;
  int n = tid >> 6, j = tid & 63;
  float di = dinv[n];
  agg[tid] = xt[tid] * di * di + cb[j];
}

// ----------------------------------------------- edge scatter (bandwidth)
__global__ __launch_bounds__(256) void edge_agg_kernel(
    const int* __restrict__ src, const int* __restrict__ dst,
    const float* __restrict__ dinv, const float* __restrict__ xt,
    float* __restrict__ agg) {
  int tid = blockIdx.x * 256 + threadIdx.x;      // E*64 threads
  int e = tid >> 6, j = tid & 63;
  int s = src[e], d = dst[e];
  float en = dinv[s] * dinv[d];
  atomicAdd(&agg[(size_t)d * HID + j], xt[(size_t)s * HID + j] * en);
}

// --------------------------------------- layernorm + relu + residual
__global__ __launch_bounds__(64) void ln_kernel(
    const float* __restrict__ agg, float* __restrict__ h,
    const float* __restrict__ g, const float* __restrict__ b) {
  __shared__ float sd[64];
  int n = blockIdx.x, t = threadIdx.x;
  float a = agg[(size_t)n * HID + t];
  sd[t] = a;
  __syncthreads();
  for (int o = 32; o > 0; o >>= 1) { if (t < o) sd[t] += sd[t + o]; __syncthreads(); }
  float mu = sd[0] * (1.0f / HID);
  __syncthreads();
  float dvv = a - mu;
  sd[t] = dvv * dvv;
  __syncthreads();
  for (int o = 32; o > 0; o >>= 1) { if (t < o) sd[t] += sd[t + o]; __syncthreads(); }
  float var = sd[0] * (1.0f / HID);
  float hn = dvv * rsqrtf(var + 1e-5f) * g[t] + b[t];
  size_t idx = (size_t)n * HID + t;
  h[idx] = fmaxf(hn, 0.0f) + h[idx];   // residual (identity read, then overwrite)
}

// ------------------------------------------------ LSTM + FC head (WMMA)
// LDS: [0,64KB)  = Wih+Whh B-fragments (reused as f32 hT staging after steps)
//      [64,72KB) = per-wave h_prev (bf16 row-major, 2KB each)
__global__ __launch_bounds__(128) void lstm_kernel(
    const float* __restrict__ hfin, const float* __restrict__ bsum,
    const __bf16* __restrict__ wFrag,   // wihF followed by whhF, 32768 bf16
    const float* __restrict__ fc1W, const float* __restrict__ fc1b,
    const float* __restrict__ fc2W, const float* __restrict__ fc2b,
    float* __restrict__ out) {
  __shared__ __align__(16) char smem[64 * 1024 + 4 * 16 * HID * 2];
  int w = threadIdx.x >> 5, lane = threadIdx.x & 31;

  // stage 64KB of weight fragments into LDS (once per block)
  {
    const uint4* g = (const uint4*)wFrag;
    uint4* l = (uint4*)smem;
    for (int i = threadIdx.x; i < 4096; i += 128) l[i] = g[i];
  }
  __bf16* hbuf = (__bf16*)(smem + 64 * 1024) + w * 16 * HID;
  for (int i = lane; i < 16 * HID; i += 32) hbuf[i] = (__bf16)0.0f;
  __syncthreads();

  int tile = blockIdx.x * 4 + w;
  bool valid = tile < NTILES;
  if (!valid) tile = NTILES - 1;
  int base = tile * 16, rlo = lane & 15, hi8 = (lane >= 16) ? 8 : 0;
  const v16bf* wihL = (const v16bf*)smem;          // 1024 fragments
  const v16bf* whhL = wihL + 1024;

  v8f cpr[4], hreg[4];
#pragma unroll
  for (int j = 0; j < 4; ++j)
#pragma unroll
    for (int r = 0; r < 8; ++r) { cpr[j][r] = 0.0f; hreg[j][r] = 0.0f; }

  for (int t = 0; t < SEQ; ++t) {
    // A_x: row (node - 9 + t) of GNN output, zero-masked
    v16bf ax0, ax1;
    int inIdx = base + rlo - (SEQ - 1) + t;
    if (inIdx >= 0) {
      const float* rp = hfin + (size_t)inIdx * HID;
#pragma unroll
      for (int i = 0; i < 8; ++i) {
        ax0[i]     = (__bf16)rp[2 * hi8 + i];
        ax0[i + 8] = (__bf16)rp[2 * hi8 + 8 + i];
        ax1[i]     = (__bf16)rp[32 + 2 * hi8 + i];
        ax1[i + 8] = (__bf16)rp[32 + 2 * hi8 + 8 + i];
      }
    } else {
#pragma unroll
      for (int i = 0; i < 16; ++i) { ax0[i] = (__bf16)0.0f; ax1[i] = (__bf16)0.0f; }
    }
    if (t < SEQ - 1 && inIdx + 1 >= 0)            // hide next step's gather
      __builtin_prefetch(hfin + (size_t)(inIdx + 1) * HID, 0, 1);

    // A_h from per-wave LDS (row-major: two contiguous 16B reads per half)
    v16bf ah0, ah1;
    const __bf16* hp = &hbuf[rlo * HID];
#pragma unroll
    for (int i = 0; i < 8; ++i) {
      ah0[i]     = hp[2 * hi8 + i];
      ah0[i + 8] = hp[2 * hi8 + 8 + i];
      ah1[i]     = hp[32 + 2 * hi8 + i];
      ah1[i + 8] = hp[32 + 2 * hi8 + 8 + i];
    }
#pragma unroll
    for (int j = 0; j < 4; ++j) {
      int col = j * 16 + rlo;
      v8f acc[4];
#pragma unroll
      for (int gidx = 0; gidx < 4; ++gidx) {       // gates i,f,g,o
        int tb = (gidx * 4 + j) * 64;              // fragment base (v16bf units)
        float bv = bsum[gidx * 64 + col];          // bias folded into C init
        v8f c;
#pragma unroll
        for (int r = 0; r < 8; ++r) c[r] = bv;
        c = wmma_bf16(ax0, wihL[tb + lane], c);
        c = wmma_bf16(ax1, wihL[tb + 32 + lane], c);
        c = wmma_bf16(ah0, whhL[tb + lane], c);
        c = wmma_bf16(ah1, whhL[tb + 32 + lane], c);
        acc[gidx] = c;
      }
#pragma unroll
      for (int r = 0; r < 8; ++r) {
        float ig = sigf(acc[0][r]);
        float fg = sigf(acc[1][r]);
        float gg = tanhf(acc[2][r]);
        float og = sigf(acc[3][r]);
        float cv = fg * cpr[j][r] + ig * gg;
        cpr[j][r] = cv;
        float hv = og * tanhf(cv);
        hreg[j][r] = hv;
        hbuf[(r + hi8) * HID + col] = (__bf16)hv;  // h_prev for next step
      }
    }
  }

  // weights are dead now; reuse their LDS for the f32 hT staging
  __syncthreads();
  float* hTw = (float*)smem + w * 16 * HID;
#pragma unroll
  for (int j = 0; j < 4; ++j)
#pragma unroll
    for (int r = 0; r < 8; ++r)
      hTw[(r + hi8) * HID + j * 16 + rlo] = hreg[j][r];

  // FC head: out = relu(hT @ fc1^T + b1) @ fc2^T + b2
  int mh = lane >> 4;               // lane half handles 16 of the 32 fc1 rows
  float s2 = 0.0f;
  const float* hrow = hTw + rlo * HID;
  for (int m = 0; m < 16; ++m) {
    int mi = mh * 16 + m;
    float o1 = fc1b[mi];
    const float* wr = fc1W + mi * HID;
#pragma unroll
    for (int k = 0; k < HID; ++k) o1 += hrow[k] * wr[k];
    s2 += fmaxf(o1, 0.0f) * fc2W[mi];
  }
  s2 += __shfl_xor(s2, 16, 32);
  if (valid && lane < 16) out[base + rlo] = s2 + fc2b[0];
}

// ----------------------------------------------------------------- launch
extern "C" void kernel_launch(void* const* d_in, const int* in_sizes, int n_in,
                              void* d_out, int out_size, void* d_ws, size_t ws_size,
                              hipStream_t stream) {
  const float* x     = (const float*)d_in[0];
  const int*   ei    = (const int*)d_in[1];     // [2,E]: row0=src, row1=dst
  const float* Win   = (const float*)d_in[2];
  const float* bin   = (const float*)d_in[3];
  const float* convW = (const float*)d_in[4];
  const float* convb = (const float*)d_in[5];
  const float* lng   = (const float*)d_in[6];
  const float* lnb   = (const float*)d_in[7];
  const float* Wih   = (const float*)d_in[8];
  const float* Whh   = (const float*)d_in[9];
  const float* bih   = (const float*)d_in[10];
  const float* bhh   = (const float*)d_in[11];
  const float* fc1W  = (const float*)d_in[12];
  const float* fc1b  = (const float*)d_in[13];
  const float* fc2W  = (const float*)d_in[14];
  const float* fc2b  = (const float*)d_in[15];
  float* out = (float*)d_out;

  float* deg  = (float*)d_ws;
  float* dinv = deg + N_NODES;
  float* h    = dinv + N_NODES;
  float* xt   = h  + (size_t)N_NODES * HID;
  float* agg  = xt + (size_t)N_NODES * HID;
  __bf16* wihF  = (__bf16*)(agg + (size_t)N_NODES * HID);
  __bf16* whhF  = wihF + 16 * 2 * 32 * 16;     // contiguous after wihF
  __bf16* convF = whhF + 16 * 2 * 32 * 16;
  float*  bsum  = (float*)(convF + 2 * 4 * 2 * 32 * 16);

  const int* src = ei;
  const int* dst = ei + N_EDGES;

  hipMemsetAsync(deg, 0, N_NODES * sizeof(float), stream);
  deg_kernel<<<(N_EDGES + 255) / 256, 256, 0, stream>>>(dst, deg);
  dinv_kernel<<<(N_NODES + 255) / 256, 256, 0, stream>>>(deg, dinv);
  prep_kernel<<<11, 256, 0, stream>>>(Wih, Whh, convW, bih, bhh,
                                      wihF, whhF, convF, bsum);
  inproj_kernel<<<(N_NODES * HID) / 256, 256, 0, stream>>>(x, Win, bin, h);

  for (int l = 0; l < 2; ++l) {
    gcn_xt_kernel<<<(NTILES + 7) / 8, 256, 0, stream>>>(
        h, convF + (size_t)l * 4 * 2 * 32 * 16, xt);
    agg_init_kernel<<<(N_NODES * HID) / 256, 256, 0, stream>>>(
        xt, dinv, convb + l * HID, agg);
    edge_agg_kernel<<<(N_EDGES * 64) / 256, 256, 0, stream>>>(
        src, dst, dinv, xt, agg);
    ln_kernel<<<N_NODES, 64, 0, stream>>>(agg, h, lng + l * HID, lnb + l * HID);
  }

  lstm_kernel<<<(NTILES + 3) / 4, 128, 0, stream>>>(
      h, bsum, wihF, fc1W, fc1b, fc2W, fc2b, out);
}
